// NLTL_TSv3_45938970198148
// MI455X (gfx1250) — matrix operand order repaired and checked by reading, not compiled, not run on hardware
//
#include <hip/hip_runtime.h>

// ---------------------------------------------------------------------------
// CDNA5 (gfx1250) implementation of the NLTL-TS forward pass.
// All dense matmuls run through v_wmma_f32_16x16x32_bf16 (bf16 A/W, f32 acc).
// Dead-code elimination: layer-3 un/bn and layer-2 bn are never consumed by
// the final logits, so those (largest) GEMMs are skipped -- outputs identical.
// Round 1: each wave now computes NT N-tiles per A fragment (A reuse in
// registers) -- cuts A-matrix HBM traffic by NT x on the dominant GEMMs.
// ---------------------------------------------------------------------------

typedef unsigned short u16;
typedef __attribute__((ext_vector_type(16))) __bf16 bf16x16;
typedef __attribute__((ext_vector_type(8)))  __bf16 bf16x8;
typedef __attribute__((ext_vector_type(8)))  float  f32x8;

#define ACT_NONE 0
#define ACT_RELU 1
#define ACT_SIG  2

__device__ __forceinline__ u16 f2bf(float f) {
  unsigned u = __builtin_bit_cast(unsigned, f);
  unsigned r = u + 0x7FFFu + ((u >> 16) & 1u);   // round-to-nearest-even
  return (u16)(r >> 16);
}

// ---------------------------------------------------------------------------
// WMMA GEMM: C[M,NdP] = act(scale * (A[M,Kp] @ WT[NdP,Kp]^T + bias) + shift)
// A bf16 row-major [Mt*16, Kp], WT bf16 row-major [NdP, Kp] (i.e. W^T),
// Kp multiple of 32, NdP multiple of 16. One wave per 16x(16*NT) C strip:
// the A fragment is loaded once per k-step and reused for NT WMMAs.
// Fragment layouts per CDNA5 ISA 7.12.2 (wave32):
//   A 16x32 bf16 : lane row = lane&15; elems 0..7 -> K = half*8+e,
//                  elems 8..15 -> K = 16 + half*8 + (e-8)
//   B 32x16 bf16 : lane col = lane&15; same K mapping as A
//   C 16x16 f32  : vgpr r -> row = r + 8*half, col = lane&15
// ---------------------------------------------------------------------------
template <int NT>
__global__ __launch_bounds__(256) void k_gemm_bf16(
    const u16* __restrict__ A, const u16* __restrict__ WT,
    const float* __restrict__ bias, const float* __restrict__ scale,
    const float* __restrict__ shift, float* __restrict__ C,
    int Mt, int Kp, int NdP, int nReal, int act)
{
  int wave = threadIdx.x >> 5;
  int lane = threadIdx.x & 31;
  int mt = blockIdx.y * 8 + wave;
  if (mt >= Mt) return;                 // uniform per-wave exit: EXEC stays all-1s
  int nt0  = blockIdx.x * NT;
  int half = lane >> 4;
  int ll   = lane & 15;

  const u16* arow = A + (size_t)(mt * 16 + ll) * Kp;
  const u16* brow[NT];
#pragma unroll
  for (int j = 0; j < NT; ++j)
    brow[j] = WT + (size_t)((nt0 + j) * 16 + ll) * Kp;

  f32x8 acc[NT];
#pragma unroll
  for (int j = 0; j < NT; ++j)
    acc[j] = (f32x8){0.f, 0.f, 0.f, 0.f, 0.f, 0.f, 0.f, 0.f};

  for (int k0 = 0; k0 < Kp; k0 += 32) {
    bf16x8 alo = *(const bf16x8*)(arow + k0 + half * 8);        // 16B load
    bf16x8 ahi = *(const bf16x8*)(arow + k0 + 16 + half * 8);   // 16B load
    bf16x16 av = __builtin_shufflevector(alo, ahi,
        0,1,2,3,4,5,6,7,8,9,10,11,12,13,14,15);
#pragma unroll
    for (int j = 0; j < NT; ++j) {
      bf16x8 blo = *(const bf16x8*)(brow[j] + k0 + half * 8);
      bf16x8 bhi = *(const bf16x8*)(brow[j] + k0 + 16 + half * 8);
      bf16x16 bv = __builtin_shufflevector(blo, bhi,
          0,1,2,3,4,5,6,7,8,9,10,11,12,13,14,15);
      acc[j] = __builtin_amdgcn_wmma_f32_16x16x32_bf16(
          false, av, false, bv, (short)0, acc[j], false, false);
    }
  }

#pragma unroll
  for (int j = 0; j < NT; ++j) {
    int n = (nt0 + j) * 16 + ll;
    if (n < nReal) {
      float bi = bias  ? bias[n]  : 0.f;
      float sc = scale ? scale[n] : 1.f;
      float sh = shift ? shift[n] : 0.f;
#pragma unroll
      for (int r = 0; r < 8; ++r) {
        float v = sc * (acc[j][r] + bi) + sh;
        if (act == ACT_RELU)      v = fmaxf(v, 0.f);
        else if (act == ACT_SIG)  v = 1.f / (1.f + expf(-v));
        C[(size_t)(mt * 16 + half * 8 + r) * NdP + n] = v;
      }
    }
  }
}

// dispatcher: pick NT per output width (NdP is one of 32, 64, 96, 128)
static void launch_gemm(const u16* A, const u16* WT, const float* bias,
                        const float* scale, const float* shift, float* C,
                        int Mt, int Kp, int NdP, int nReal, int act,
                        hipStream_t stream) {
  dim3 blk(256);
  unsigned gy = (unsigned)((Mt + 7) / 8);
  if (NdP == 96)
    k_gemm_bf16<6><<<dim3(1, gy), blk, 0, stream>>>(A, WT, bias, scale, shift, C, Mt, Kp, NdP, nReal, act);
  else if (NdP >= 64)
    k_gemm_bf16<4><<<dim3(NdP / 64, gy), blk, 0, stream>>>(A, WT, bias, scale, shift, C, Mt, Kp, NdP, nReal, act);
  else
    k_gemm_bf16<2><<<dim3(NdP / 32, gy), blk, 0, stream>>>(A, WT, bias, scale, shift, C, Mt, Kp, NdP, nReal, act);
}

// ---- weight pack: W[Kd,Nd] f32 row-major -> WT[NdP,Kp] bf16 (zero padded) ---
__global__ void k_packW(const float* __restrict__ W, u16* __restrict__ WT,
                        int Kd, int Nd, int Kp, int NdP) {
  int i = blockIdx.x * blockDim.x + threadIdx.x;
  long long tot = (long long)NdP * Kp;
  if (i >= tot) return;
  int k = i % Kp, n = i / Kp;
  float v = (n < Nd && k < Kd) ? W[(size_t)k * Nd + n] : 0.f;
  WT[i] = f2bf(v);
}

// ---- generic f32 -> bf16 pack with column offset + optional row broadcast ---
__global__ void k_pack_cast(u16* __restrict__ dst, int dstStride, int coff,
                            const float* __restrict__ src, int srcStride,
                            int rowDiv, int rows, int cols) {
  int i = blockIdx.x * blockDim.x + threadIdx.x;
  long long tot = (long long)rows * cols;
  if (i >= tot) return;
  int c = i % cols, r = i / cols;
  dst[(size_t)r * dstStride + coff + c] =
      f2bf(src[(size_t)(r / rowDiv) * srcStride + c]);
}

// ---- stabilize: width-5 moving average with replication padding -------------
__global__ void k_stabilize(const float* __restrict__ traj, float* __restrict__ xs) {
  int i = blockIdx.x * blockDim.x + threadIdx.x;
  if (i >= 8 * 64 * 13 * 3) return;
  int s = i % 3, r = i / 3;
  int n = r % 13; r /= 13;
  int t = r % 64; int b = r / 64;
  float acc = 0.f;
#pragma unroll
  for (int kk = 0; kk < 5; ++kk) {
    int tt = t + kk - 2;
    tt = tt < 0 ? 0 : (tt > 63 ? 63 : tt);
    acc += traj[(((size_t)b * 64 + tt) * 13 + n) * 3 + s];
  }
  xs[i] = acc * 0.2f;
}

// ---- GCN 1x1 conv (S=3 inner dim) ------------------------------------------
__global__ void k_gcn_lin(const float* __restrict__ xs, const float* __restrict__ w,
                          const float* __restrict__ bias, float* __restrict__ y) {
  int i = blockIdx.x * blockDim.x + threadIdx.x;
  if (i >= 6656 * 896) return;
  int kc = i % 896; int r = i / 896;
  const float* xr = xs + (size_t)r * 3;
  y[i] = bias[kc] + xr[0] * w[kc] + xr[1] * w[896 + kc] + xr[2] * w[1792 + kc];
}

// ---- spatial graph conv + BN1 + ReLU ---------------------------------------
__global__ void k_gcn_spatial(const float* __restrict__ y, const float* __restrict__ A,
                              const float* __restrict__ g, const float* __restrict__ b,
                              float* __restrict__ g1) {
  int i = blockIdx.x * blockDim.x + threadIdx.x;
  if (i >= 6656 * 128) return;
  int c = i % 128; int r = i / 128;
  int w = r % 13; int bt = r / 13;
  float acc = 0.f;
  for (int k = 0; k < 7; ++k)
    for (int v = 0; v < 13; ++v)
      acc += y[((size_t)(bt * 13 + v)) * 896 + k * 128 + c] * A[(k * 13 + v) * 13 + w];
  float val = g[c] * acc + b[c];
  g1[i] = fmaxf(val, 0.f);
}

// ---- im2col for temporal conv (7-tap, zero pad 3) --------------------------
__global__ void k_pack_tcn(const float* __restrict__ g1, u16* __restrict__ At) {
  int i = blockIdx.x * blockDim.x + threadIdx.x;
  if (i >= 6656 * 896) return;
  int kc = i % 896; int r = i / 896;
  int tap = kc / 128, c = kc % 128;
  int n = r % 13; int bt = r / 13;
  int t = bt % 64, b = bt / 64;
  int tt = t + tap - 3;
  float v = (tt >= 0 && tt < 64) ? g1[(((size_t)b * 64 + tt) * 13 + n) * 128 + c] : 0.f;
  At[i] = f2bf(v);
}

// ---- u = [mlp2(h) | one-hot agent type], stride 224 ------------------------
__global__ void k_build_u(const float* __restrict__ um, float* __restrict__ u) {
  int i = blockIdx.x * blockDim.x + threadIdx.x;
  if (i >= 6656 * 141) return;
  int c = i % 141; int r = i / 141;
  int n = r % 13;
  float v = (c < 128) ? um[(size_t)r * 128 + c] : ((c - 128) == n ? 1.f : 0.f);
  u[(size_t)r * 224 + c] = v;
}

// ---- pair features [h_i | h_j] -> bf16 -------------------------------------
__global__ void k_pack_pair(const float* __restrict__ h, u16* __restrict__ ap) {
  int i = blockIdx.x * blockDim.x + threadIdx.x;
  long long tot = 86528LL * 256;
  if (i >= tot) return;
  int c = i % 256; int r = i / 256;
  int j = r % 13; int ii = (r / 13) % 13; int bt = r / 169;
  float v = (c < 128) ? h[((size_t)bt * 13 + ii) * 128 + c]
                      : h[((size_t)bt * 13 + j) * 128 + (c - 128)];
  ap[i] = f2bf(v);
}

// ---- red_u: unary -> nullary (max/min over agents) -> bf16 at coff ----------
__global__ void k_red_u(const float* __restrict__ u, int Du, int uStride,
                        u16* __restrict__ dst, int dstStride, int coff) {
  int i = blockIdx.x * blockDim.x + threadIdx.x;
  if (i >= 512 * Du) return;
  int c = i % Du; int bt = i / Du;
  float mx = -1e30f, mn = 1e30f;
  for (int n = 0; n < 13; ++n) {
    float v = u[((size_t)(bt * 13 + n)) * uStride + c];
    mx = fmaxf(mx, v); mn = fminf(mn, v);
  }
  dst[(size_t)bt * dstStride + coff + c]      = f2bf(mx);
  dst[(size_t)bt * dstStride + coff + Du + c] = f2bf(mn);
}

// ---- red_b: binary -> unary, exclude self (diag -> 0 for max, 1 for min) ----
__global__ void k_red_b(const float* __restrict__ b, int Db, int bStride,
                        u16* __restrict__ dst, int dstStride, int coff) {
  int i = blockIdx.x * blockDim.x + threadIdx.x;
  long long tot = 6656LL * Db;
  if (i >= tot) return;
  int c = i % Db; int r = i / Db;
  int ii = r % 13; int bt = r / 13;
  float mx = -1e30f, mn = 1e30f;
  for (int j = 0; j < 13; ++j) {
    float vx, vn;
    if (j == ii) { vx = 0.f; vn = 1.f; }
    else {
      float v = b[((size_t)bt * 169 + ii * 13 + j) * bStride + c];
      vx = v; vn = v;
    }
    mx = fmaxf(mx, vx); mn = fminf(mn, vn);
  }
  dst[(size_t)r * dstStride + coff + c]      = f2bf(mx);
  dst[(size_t)r * dstStride + coff + Db + c] = f2bf(mn);
}

// ---- f2 = [[u_i|b_ij] , [u_j|b_ji]] -> bf16 --------------------------------
__global__ void k_pack_f2(const float* __restrict__ u, int Du, int uS,
                          const float* __restrict__ b, int Db, int bS,
                          u16* __restrict__ dst, int dstStride) {
  int W2 = 2 * (Du + Db);
  long long tot = 86528LL * W2;
  long long i = (long long)blockIdx.x * blockDim.x + threadIdx.x;
  if (i >= tot) return;
  int c = (int)(i % W2); int r = (int)(i / W2);
  int j = r % 13; int ii = (r / 13) % 13; int bt = r / 169;
  float v;
  if (c < Du)                 v = u[((size_t)bt * 13 + ii) * uS + c];
  else if (c < Du + Db)       v = b[(size_t)r * bS + (c - Du)];
  else if (c < 2 * Du + Db)   v = u[((size_t)bt * 13 + j) * uS + (c - Du - Db)];
  else                        v = b[((size_t)bt * 169 + j * 13 + ii) * bS + (c - 2 * Du - Db)];
  dst[(size_t)r * dstStride + c] = f2bf(v);
}

// ---- residual concat dst = [s0 | s1] ---------------------------------------
__global__ void k_concat2(float* __restrict__ dst, int dS,
                          const float* __restrict__ s0, int s0S, int d0,
                          const float* __restrict__ s1, int s1S, int d1,
                          int rows) {
  int W2 = d0 + d1;
  long long tot = (long long)rows * W2;
  long long i = (long long)blockIdx.x * blockDim.x + threadIdx.x;
  if (i >= tot) return;
  int c = (int)(i % W2); int r = (int)(i / W2);
  dst[(size_t)r * dS + c] = (c < d0) ? s0[(size_t)r * s0S + c]
                                     : s1[(size_t)r * s1S + (c - d0)];
}

__global__ void k_copy_strided(float* __restrict__ dst, int dS,
                               const float* __restrict__ src, int sS,
                               int rows, int cols) {
  long long tot = (long long)rows * cols;
  long long i = (long long)blockIdx.x * blockDim.x + threadIdx.x;
  if (i >= tot) return;
  int c = (int)(i % cols); int r = (int)(i / cols);
  dst[(size_t)r * dS + c] = src[(size_t)r * sS + c];
}

// ---- TLM: build [x|cummax_f|cummax_b|cummin_f|cummin_b] in bf16 ------------
__global__ void k_tlm_pack(const float* __restrict__ x, u16* __restrict__ A) {
  int i = blockIdx.x * blockDim.x + threadIdx.x;
  if (i >= 8 * 96) return;
  int c = i % 96; int b = i / 96;
  float rmx = -1e30f, rmn = 1e30f;
  for (int t = 0; t < 64; ++t) {
    int row = b * 64 + t;
    float v = x[(size_t)row * 96 + c];
    rmx = fmaxf(rmx, v); rmn = fminf(rmn, v);
    A[(size_t)row * 480 + c]       = f2bf(v);
    A[(size_t)row * 480 + 96 + c]  = f2bf(rmx);
    A[(size_t)row * 480 + 288 + c] = f2bf(rmn);
  }
  rmx = -1e30f; rmn = 1e30f;
  for (int t = 63; t >= 0; --t) {
    int row = b * 64 + t;
    float v = x[(size_t)row * 96 + c];
    rmx = fmaxf(rmx, v); rmn = fminf(rmn, v);
    A[(size_t)row * 480 + 192 + c] = f2bf(rmx);
    A[(size_t)row * 480 + 384 + c] = f2bf(rmn);
  }
}

__global__ void k_feat_max(const float* __restrict__ x, u16* __restrict__ Adec) {
  int i = blockIdx.x * blockDim.x + threadIdx.x;
  if (i >= 8 * 96) return;
  int c = i % 96; int b = i / 96;
  float mx = -1e30f;
  for (int t = 0; t < 64; ++t) mx = fmaxf(mx, x[((size_t)b * 64 + t) * 96 + c]);
  Adec[(size_t)b * 96 + c] = f2bf(mx);
}

__global__ void k_out(const float* __restrict__ Cdec, float* __restrict__ out) {
  int i = blockIdx.x * blockDim.x + threadIdx.x;
  if (i >= 8 * 19) return;
  int n = i % 19; int b = i / 19;
  out[i] = Cdec[(size_t)b * 32 + n];
}

// ---------------------------------------------------------------------------
static inline unsigned gb(long long n) { return (unsigned)((n + 255) / 256); }

extern "C" void kernel_launch(void* const* d_in, const int* in_sizes, int n_in,
                              void* d_out, int out_size, void* d_ws, size_t ws_size,
                              hipStream_t stream) {
  // ---- inputs (setup_inputs dict order) ----
  const float* traj  = (const float*)d_in[0];
  const float* Aadj  = (const float*)d_in[2];
  const float* w_gcn = (const float*)d_in[3];
  const float* b_gcn = (const float*)d_in[4];
  const float* bn1g  = (const float*)d_in[5];
  const float* bn1b  = (const float*)d_in[6];
  const float* w_tcn = (const float*)d_in[7];
  const float* b_tcn = (const float*)d_in[8];
  const float* bn2g  = (const float*)d_in[9];
  const float* bn2b  = (const float*)d_in[10];
  const float* w_u1  = (const float*)d_in[11];
  const float* b_u1  = (const float*)d_in[12];
  const float* w_u2  = (const float*)d_in[13];
  const float* b_u2  = (const float*)d_in[14];
  const float* w_b1  = (const float*)d_in[15];
  const float* b_b1  = (const float*)d_in[16];
  const float* w_b2  = (const float*)d_in[17];
  const float* b_b2  = (const float*)d_in[18];
  const float* nb1   = (const float*)d_in[19];
  const float* nw2   = (const float*)d_in[20];
  const float* nb2   = (const float*)d_in[21];
  const float* tlm_w = (const float*)d_in[22];
  const float* tlm_b = (const float*)d_in[23];
  const float* dec_w = (const float*)d_in[24];
  const float* dec_b = (const float*)d_in[25];
  const float* nw_10 = (const float*)d_in[26];
  const float* nw_11 = (const float*)d_in[27];
  const float* nw_12 = (const float*)d_in[28];
  const float* nw_20 = (const float*)d_in[29];
  const float* nw_21 = (const float*)d_in[30];
  const float* nw_30 = (const float*)d_in[32];   // nw_22(31), nw_31(33), nw_32(34) are dead
  float* out = (float*)d_out;

  // ---- workspace bump allocator ----
  char* base = (char*)d_ws;
  size_t off = 0;
  auto alloc = [&](size_t bytes) -> void* {
    off = (off + 255) & ~(size_t)255;
    void* p = base + off;
    off += bytes;
    return p;
  };
  const size_t MU = 6656, MB2 = 86528, BT = 512;

  float* xs    = (float*)alloc(19968 * 4);
  float* y_gcn = (float*)alloc(MU * 896 * 4);
  float* g1    = (float*)alloc(MU * 128 * 4);
  u16*   A_tcn = (u16*)  alloc(MU * 896 * 2);
  float* h     = (float*)alloc(MU * 128 * 4);
  u16*   A_h   = (u16*)  alloc(MU * 128 * 2);
  float* t1    = (float*)alloc(MU * 128 * 4);
  u16*   A_t1  = (u16*)  alloc(MU * 128 * 2);
  float* u_mlp = (float*)alloc(MU * 128 * 4);
  float* u_a   = (float*)alloc(MU * 224 * 4);
  float* u_b   = (float*)alloc(MU * 224 * 4);
  u16*   A_pair= (u16*)  alloc(MB2 * 256 * 2);
  float* t2    = (float*)alloc(MB2 * 128 * 4);
  u16*   A_t2  = (u16*)  alloc(MB2 * 128 * 2);
  float* b_a   = (float*)alloc(MB2 * 128 * 4);
  float* b_b   = (float*)alloc(MB2 * 160 * 4);
  u16*   A_f0  = (u16*)  alloc(BT * 480 * 2);
  u16*   A_f1  = (u16*)  alloc(MU * 544 * 2);
  u16*   A_f2  = (u16*)  alloc(MB2 * 544 * 2);
  u16*   A_mn  = (u16*)  alloc(BT * 64 * 2);
  u16*   A_mu  = (u16*)  alloc(MU * 64 * 2);
  u16*   A_mb  = (u16*)  alloc(MB2 * 64 * 2);
  float* hid_n = (float*)alloc(BT * 64 * 4);
  float* hid_u = (float*)alloc(MU * 64 * 4);
  float* hid_b = (float*)alloc(MB2 * 64 * 4);
  float* o_n   = (float*)alloc(BT * 32 * 4);
  float* o_u   = (float*)alloc(MU * 32 * 4);
  float* o_b   = (float*)alloc(MB2 * 32 * 4);
  float* n0_a  = (float*)alloc(BT * 96 * 4);
  float* n0_b  = (float*)alloc(BT * 96 * 4);
  float* x_a   = (float*)alloc(BT * 96 * 4);
  float* x_b   = (float*)alloc(BT * 96 * 4);
  u16*   A_tlm = (u16*)  alloc(BT * 480 * 2);
  u16*   A_dec = (u16*)  alloc(16 * 96 * 2);
  float* Cdec  = (float*)alloc(16 * 32 * 4);
  u16*   WT_tcn= (u16*)  alloc(128 * 896 * 2);
  u16*   WT_u1 = (u16*)  alloc(128 * 128 * 2);
  u16*   WT_u2 = (u16*)  alloc(128 * 128 * 2);
  u16*   WT_b1 = (u16*)  alloc(128 * 256 * 2);
  u16*   WT_b2 = (u16*)  alloc(128 * 128 * 2);
  u16*   WT_s0 = (u16*)  alloc(64 * 480 * 2);
  u16*   WT_s1 = (u16*)  alloc(64 * 544 * 2);
  u16*   WT_s2 = (u16*)  alloc(64 * 544 * 2);
  u16*   WT_s3 = (u16*)  alloc(64 * 480 * 2);
  u16*   WT_s4 = (u16*)  alloc(64 * 544 * 2);
  u16*   WT_s6 = (u16*)  alloc(64 * 480 * 2);
  u16*   WT2[7];
  for (int i = 0; i < 7; ++i) WT2[i] = (u16*)alloc(32 * 64 * 2);
  u16*   WT_tlm[3];
  for (int i = 0; i < 3; ++i) WT_tlm[i] = (u16*)alloc(96 * 480 * 2);
  u16*   WT_dec= (u16*)  alloc(32 * 96 * 2);

  dim3 B256(256);
  // ---- pack all weights (padded K, transposed, bf16) ----
  k_packW<<<gb(128LL*896), B256, 0, stream>>>(w_tcn, WT_tcn, 896, 128, 896, 128);
  k_packW<<<gb(128LL*128), B256, 0, stream>>>(w_u1, WT_u1, 128, 128, 128, 128);
  k_packW<<<gb(128LL*128), B256, 0, stream>>>(w_u2, WT_u2, 128, 128, 128, 128);
  k_packW<<<gb(128LL*256), B256, 0, stream>>>(w_b1, WT_b1, 256, 128, 256, 128);
  k_packW<<<gb(128LL*128), B256, 0, stream>>>(w_b2, WT_b2, 128, 128, 128, 128);
  k_packW<<<gb(64LL*480), B256, 0, stream>>>(nw_10, WT_s0, 282, 64, 480, 64);
  k_packW<<<gb(64LL*544), B256, 0, stream>>>(nw_11, WT_s1, 397, 64, 544, 64);
  k_packW<<<gb(64LL*544), B256, 0, stream>>>(nw_12, WT_s2, 538, 64, 544, 64);
  k_packW<<<gb(64LL*480), B256, 0, stream>>>(nw_20, WT_s3, 378, 64, 480, 64);
  k_packW<<<gb(64LL*544), B256, 0, stream>>>(nw_21, WT_s4, 525, 64, 544, 64);
  k_packW<<<gb(64LL*480), B256, 0, stream>>>(nw_30, WT_s6, 474, 64, 480, 64);
  const int used_idx[6] = {0, 1, 2, 3, 4, 6};
  for (int s = 0; s < 6; ++s) {
    int idx = used_idx[s];
    k_packW<<<gb(32LL*64), B256, 0, stream>>>(nw2 + (size_t)idx * 64 * 32, WT2[idx], 64, 32, 64, 32);
  }
  for (int i = 0; i < 3; ++i)
    k_packW<<<gb(96LL*480), B256, 0, stream>>>(tlm_w + (size_t)i * 480 * 96, WT_tlm[i], 480, 96, 480, 96);
  k_packW<<<gb(32LL*96), B256, 0, stream>>>(dec_w, WT_dec, 96, 19, 96, 32);

  // ---- zero the K-padded concat buffers once ----
  hipMemsetAsync(A_f0, 0, BT * 480 * 2, stream);
  hipMemsetAsync(A_f1, 0, MU * 544 * 2, stream);
  hipMemsetAsync(A_f2, 0, MB2 * 544 * 2, stream);
  hipMemsetAsync(A_dec, 0, 16 * 96 * 2, stream);

  // ---- STGCN ----
  k_stabilize<<<gb(19968), B256, 0, stream>>>(traj, xs);
  k_gcn_lin<<<gb((long long)MU*896), B256, 0, stream>>>(xs, w_gcn, b_gcn, y_gcn);
  k_gcn_spatial<<<gb((long long)MU*128), B256, 0, stream>>>(y_gcn, Aadj, bn1g, bn1b, g1);
  k_pack_tcn<<<gb((long long)MU*896), B256, 0, stream>>>(g1, A_tcn);
  launch_gemm(A_tcn, WT_tcn, b_tcn, bn2g, bn2b, h, 416, 896, 128, 128, ACT_RELU, stream);

  // ---- unary MLP2 -> u ----
  k_pack_cast<<<gb((long long)MU*128), B256, 0, stream>>>(A_h, 128, 0, h, 128, 1, (int)MU, 128);
  launch_gemm(A_h, WT_u1, b_u1, nullptr, nullptr, t1, 416, 128, 128, 128, ACT_RELU, stream);
  k_pack_cast<<<gb((long long)MU*128), B256, 0, stream>>>(A_t1, 128, 0, t1, 128, 1, (int)MU, 128);
  launch_gemm(A_t1, WT_u2, b_u2, nullptr, nullptr, u_mlp, 416, 128, 128, 128, ACT_SIG, stream);
  k_build_u<<<gb((long long)MU*141), B256, 0, stream>>>(u_mlp, u_a);

  // ---- binary MLP2 -> b ----
  k_pack_pair<<<gb((long long)MB2*256), B256, 0, stream>>>(h, A_pair);
  launch_gemm(A_pair, WT_b1, b_b1, nullptr, nullptr, t2, 5408, 256, 128, 128, ACT_RELU, stream);
  k_pack_cast<<<gb((long long)MB2*128), B256, 0, stream>>>(A_t2, 128, 0, t2, 128, 1, (int)MB2, 128);
  launch_gemm(A_t2, WT_b2, b_b2, nullptr, nullptr, b_a, 5408, 128, 128, 128, ACT_SIG, stream);

  // ================= NLM layer 1 (n0=None, Du=141, Db=128) =================
  // f0 branch (idx 0)
  k_red_u<<<gb(512LL*141), B256, 0, stream>>>(u_a, 141, 224, A_f0, 480, 0);
  launch_gemm(A_f0, WT_s0, nb1 + 0, nullptr, nullptr, hid_n, 32, 480, 64, 64, ACT_RELU, stream);
  k_pack_cast<<<gb(512LL*64), B256, 0, stream>>>(A_mn, 64, 0, hid_n, 64, 1, (int)BT, 64);
  launch_gemm(A_mn, WT2[0], nb2 + 0, nullptr, nullptr, o_n, 32, 64, 32, 32, ACT_SIG, stream);
  // f1 branch (idx 1)
  k_pack_cast<<<gb((long long)MU*141), B256, 0, stream>>>(A_f1, 544, 0, u_a, 224, 1, (int)MU, 141);
  k_red_b<<<gb((long long)MU*128), B256, 0, stream>>>(b_a, 128, 128, A_f1, 544, 141);
  launch_gemm(A_f1, WT_s1, nb1 + 64, nullptr, nullptr, hid_u, 416, 544, 64, 64, ACT_RELU, stream);
  k_pack_cast<<<gb((long long)MU*64), B256, 0, stream>>>(A_mu, 64, 0, hid_u, 64, 1, (int)MU, 64);
  launch_gemm(A_mu, WT2[1], nb2 + 32, nullptr, nullptr, o_u, 416, 64, 32, 32, ACT_SIG, stream);
  // f2 branch (idx 2)
  k_pack_f2<<<gb((long long)MB2*538), B256, 0, stream>>>(u_a, 141, 224, b_a, 128, 128, A_f2, 544);
  launch_gemm(A_f2, WT_s2, nb1 + 128, nullptr, nullptr, hid_b, 5408, 544, 64, 64, ACT_RELU, stream);
  k_pack_cast<<<gb((long long)MB2*64), B256, 0, stream>>>(A_mb, 64, 0, hid_b, 64, 1, (int)MB2, 64);
  launch_gemm(A_mb, WT2[2], nb2 + 64, nullptr, nullptr, o_b, 5408, 64, 32, 32, ACT_SIG, stream);
  // residual updates
  k_copy_strided<<<gb(512LL*32), B256, 0, stream>>>(n0_a, 96, o_n, 32, (int)BT, 32);
  k_concat2<<<gb((long long)MU*173), B256, 0, stream>>>(u_b, 224, o_u, 32, 32, u_a, 224, 141, (int)MU);
  k_concat2<<<gb((long long)MB2*160), B256, 0, stream>>>(b_b, 160, o_b, 32, 32, b_a, 128, 128, (int)MB2);

  // ================= NLM layer 2 (Dn=32, Du=173, Db=160) ===================
  // f0 branch (idx 3)
  k_pack_cast<<<gb(512LL*32), B256, 0, stream>>>(A_f0, 480, 0, n0_a, 96, 1, (int)BT, 32);
  k_red_u<<<gb(512LL*173), B256, 0, stream>>>(u_b, 173, 224, A_f0, 480, 32);
  launch_gemm(A_f0, WT_s3, nb1 + 192, nullptr, nullptr, hid_n, 32, 480, 64, 64, ACT_RELU, stream);
  k_pack_cast<<<gb(512LL*64), B256, 0, stream>>>(A_mn, 64, 0, hid_n, 64, 1, (int)BT, 64);
  launch_gemm(A_mn, WT2[3], nb2 + 96, nullptr, nullptr, o_n, 32, 64, 32, 32, ACT_SIG, stream);
  // f1 branch (idx 4)
  k_pack_cast<<<gb((long long)MU*32), B256, 0, stream>>>(A_f1, 544, 0, n0_a, 96, 13, (int)MU, 32);
  k_pack_cast<<<gb((long long)MU*173), B256, 0, stream>>>(A_f1, 544, 32, u_b, 224, 1, (int)MU, 173);
  k_red_b<<<gb((long long)MU*160), B256, 0, stream>>>(b_b, 160, 160, A_f1, 544, 205);
  launch_gemm(A_f1, WT_s4, nb1 + 256, nullptr, nullptr, hid_u, 416, 544, 64, 64, ACT_RELU, stream);
  k_pack_cast<<<gb((long long)MU*64), B256, 0, stream>>>(A_mu, 64, 0, hid_u, 64, 1, (int)MU, 64);
  launch_gemm(A_mu, WT2[4], nb2 + 128, nullptr, nullptr, o_u, 416, 64, 32, 32, ACT_SIG, stream);
  // (f2 branch dead: bn of layer 2 never consumed by live outputs)
  k_concat2<<<gb(512LL*64), B256, 0, stream>>>(n0_b, 96, o_n, 32, 32, n0_a, 96, 32, (int)BT);
  k_concat2<<<gb((long long)MU*205), B256, 0, stream>>>(u_a, 224, o_u, 32, 32, u_b, 224, 173, (int)MU);

  // ================= NLM layer 3 (Dn=64, Du=205) ===========================
  // f0 branch (idx 6); f1/f2 dead
  k_pack_cast<<<gb(512LL*64), B256, 0, stream>>>(A_f0, 480, 0, n0_b, 96, 1, (int)BT, 64);
  k_red_u<<<gb(512LL*205), B256, 0, stream>>>(u_a, 205, 224, A_f0, 480, 64);
  launch_gemm(A_f0, WT_s6, nb1 + 384, nullptr, nullptr, hid_n, 32, 480, 64, 64, ACT_RELU, stream);
  k_pack_cast<<<gb(512LL*64), B256, 0, stream>>>(A_mn, 64, 0, hid_n, 64, 1, (int)BT, 64);
  launch_gemm(A_mn, WT2[6], nb2 + 192, nullptr, nullptr, o_n, 32, 64, 32, 32, ACT_SIG, stream);
  k_concat2<<<gb(512LL*96), B256, 0, stream>>>(n0_a, 96, o_n, 32, 32, n0_b, 96, 64, (int)BT);

  // ================= Temporal Logic Machine (3 DP layers) ==================
  float* cur = n0_a;
  float* xbuf[2] = {x_a, x_b};
  for (int i = 0; i < 3; ++i) {
    k_tlm_pack<<<gb(8LL*96), B256, 0, stream>>>(cur, A_tlm);
    float* dst = xbuf[i & 1];
    launch_gemm(A_tlm, WT_tlm[i], tlm_b + 96 * i, nullptr, nullptr, dst, 32, 480, 96, 96, ACT_SIG, stream);
    cur = dst;
  }

  // ---- max over time + decoder ----
  k_feat_max<<<gb(8LL*96), B256, 0, stream>>>(cur, A_dec);
  launch_gemm(A_dec, WT_dec, dec_b, nullptr, nullptr, Cdec, 1, 96, 32, 19, ACT_NONE, stream);
  k_out<<<gb(152), B256, 0, stream>>>(Cdec, out);

  (void)in_sizes; (void)n_in; (void)out_size; (void)ws_size;
}
